// KAconv_61684320305433
// MI455X (gfx1250) — matrix operand order repaired and checked by reading, not compiled, not run on hardware
//
#include <hip/hip_runtime.h>
#include <math.h>

typedef __attribute__((ext_vector_type(16))) _Float16 v16h;
typedef __attribute__((ext_vector_type(8)))  _Float16 v8h;
typedef __attribute__((ext_vector_type(8)))  float    v8f;

#define B_   2
#define C_   32
#define H_   64
#define W_   64
#define P_   4096      // H_*W_
#define F_   288       // C_*9
#define K_   32
#define NB_  32
#define HID_ 64
#define CO_  64
#define COLS_ 18432    // CO_*F_

// ---------------------------------------------------------------------------
// 1) unfold 3x3 (zero pad) -> patches f32 + f16
// ---------------------------------------------------------------------------
__global__ void unfold_kernel(const float* __restrict__ x,
                              float* __restrict__ patf,
                              _Float16* __restrict__ path) {
    int t = blockIdx.x * blockDim.x + threadIdx.x;
    if (t >= B_ * P_ * F_) return;
    int f  = t % F_;
    int bp = t / F_;
    int p  = bp % P_;
    int b  = bp / P_;
    int c   = f / 9;
    int tap = f % 9;
    int di = tap / 3 - 1, dj = tap % 3 - 1;
    int h = p >> 6, w = p & 63;
    int hh = h + di, ww = w + dj;
    float v = 0.f;
    if (hh >= 0 && hh < H_ && ww >= 0 && ww < W_)
        v = x[(((size_t)b * C_ + c) * H_ + hh) * W_ + ww];
    patf[t] = v;
    path[t] = (_Float16)v;
}

// ---------------------------------------------------------------------------
// 2) obs[b,p,c] = mean over 9 taps
// ---------------------------------------------------------------------------
__global__ void obs_kernel(const float* __restrict__ patf, float* __restrict__ obs) {
    int t = blockIdx.x * blockDim.x + threadIdx.x;
    if (t >= B_ * P_ * C_) return;
    int c  = t % C_;
    int bp = t / C_;
    const float* pr = patf + (size_t)bp * F_ + c * 9;
    float s = 0.f;
#pragma unroll
    for (int i = 0; i < 9; ++i) s += pr[i];
    obs[t] = s * (1.0f / 9.0f);
}

// ---------------------------------------------------------------------------
// 3) k-means: 1 block per batch, 1024 threads. Deterministic (no atomics):
//    thread (k,d) owns one accumulator and scans all P in fixed order.
// ---------------------------------------------------------------------------
__global__ __launch_bounds__(1024)
void kmeans_kernel(const float* __restrict__ obs, int* __restrict__ idx_out) {
    __shared__ float cen[K_][C_];
    __shared__ float cnorm[K_];
    __shared__ unsigned char a[P_];
    __shared__ float red[1024];
    __shared__ float knorm[K_];
    __shared__ int sflag;

    const int b = blockIdx.x;
    const int t = threadIdx.x;
    const float* O = obs + (size_t)b * P_ * C_;

    // deterministic init: strided samples (reference uses a random permutation;
    // k-means is an unsupervised clustering - any fixed seed choice is valid here)
    if (t < K_ * C_) cen[t >> 5][t & 31] = O[(size_t)(t >> 5) * (P_ / K_) * C_ + (t & 31)];
    __syncthreads();

    for (int it = 0; it < 20; ++it) {
        if (t < K_) {
            float s = 0.f;
#pragma unroll
            for (int d = 0; d < C_; ++d) s += cen[t][d] * cen[t][d];
            cnorm[t] = s;
        }
        __syncthreads();
        // assignment
        for (int p = t; p < P_; p += 1024) {
            const float* row = O + (size_t)p * C_;
            float r[C_];
#pragma unroll
            for (int d = 0; d < C_; ++d) r[d] = row[d];
            int best = 0; float bv = 3.0e38f;
            for (int k = 0; k < K_; ++k) {
                float dot = 0.f;
#pragma unroll
                for (int d = 0; d < C_; ++d) dot += r[d] * cen[k][d];
                float sc = cnorm[k] - 2.0f * dot;   // |s|^2 constant: same argmin
                if (sc < bv) { bv = sc; best = k; }
            }
            a[p] = (unsigned char)best;
        }
        __syncthreads();
        // deterministic center update: thread t -> (k = t>>5, d = t&31)
        const int k = t >> 5, d = t & 31;
        float s = 0.f, c = 0.f;
        for (int p = 0; p < P_; ++p) {
            if ((int)a[p] == k) { s += O[(size_t)p * C_ + d]; c += 1.f; }
        }
        float old = cen[k][d];
        float nv  = (c > 0.f) ? (s / c) : old;
        float df  = nv - old;
        red[t] = df * df;
        cen[k][d] = nv;                 // each (k,d) owned by exactly one thread
        __syncthreads();
        if (t < K_) {
            float s2 = 0.f;
#pragma unroll
            for (int d2 = 0; d2 < C_; ++d2) s2 += red[t * C_ + d2];
            knorm[t] = sqrtf(s2);
        }
        __syncthreads();
        if (t == 0) {
            float sh = 0.f;
            for (int kk = 0; kk < K_; ++kk) sh += knorm[kk];
            sflag = (sh < 0.005f * (float)P_) ? 1 : 0;
        }
        __syncthreads();
        if (sflag) break;
    }
    // final assignment with final centers
    __syncthreads();
    if (t < K_) {
        float s = 0.f;
#pragma unroll
        for (int d = 0; d < C_; ++d) s += cen[t][d] * cen[t][d];
        cnorm[t] = s;
    }
    __syncthreads();
    for (int p = t; p < P_; p += 1024) {
        const float* row = O + (size_t)p * C_;
        float r[C_];
#pragma unroll
        for (int d = 0; d < C_; ++d) r[d] = row[d];
        int best = 0; float bv = 3.0e38f;
        for (int k = 0; k < K_; ++k) {
            float dot = 0.f;
#pragma unroll
            for (int d = 0; d < C_; ++d) dot += r[d] * cen[k][d];
            float sc = cnorm[k] - 2.0f * dot;
            if (sc < bv) { bv = sc; best = k; }
        }
        idx_out[b * P_ + p] = best;
    }
}

// ---------------------------------------------------------------------------
// 4) stable cluster ordering: offs/cnts + order (rank = #earlier same-cluster)
// ---------------------------------------------------------------------------
__global__ __launch_bounds__(1024)
void rank_kernel(const int* __restrict__ idx, int* __restrict__ order,
                 int* __restrict__ offs, int* __restrict__ cnts) {
    __shared__ int sidx[P_];
    __shared__ int scnt[K_];
    __shared__ int soff[K_];
    const int b = blockIdx.x, t = threadIdx.x;
    for (int p = t; p < P_; p += 1024) sidx[p] = idx[b * P_ + p];
    __syncthreads();
    if (t < K_) {
        int c = 0;
        for (int p = 0; p < P_; ++p) c += (sidx[p] == t);
        scnt[t] = c;
    }
    __syncthreads();
    if (t == 0) {
        int acc = 0;
        for (int k = 0; k < K_; ++k) { soff[k] = acc; acc += scnt[k]; }
    }
    __syncthreads();
    for (int p = t; p < P_; p += 1024) {
        int k = sidx[p];
        int r = 0;
        for (int q = 0; q < p; ++q) r += (sidx[q] == k);
        order[b * P_ + soff[k] + r] = p;
    }
    if (t < K_) { offs[b * K_ + t] = soff[t]; cnts[b * K_ + t] = scnt[t]; }
}

// ---------------------------------------------------------------------------
// 5) cluster centers over full patches (deterministic serial-p scan)
// ---------------------------------------------------------------------------
__global__ void centers_kernel(const int* __restrict__ idx, const float* __restrict__ patf,
                               float* __restrict__ centers) {
    const int b = blockIdx.x, k = blockIdx.y, f = threadIdx.x;
    const int* I = idx + b * P_;
    float s = 0.f, c = 0.f;
    for (int p = 0; p < P_; ++p) {
        if (I[p] == k) { s += patf[((size_t)b * P_ + p) * F_ + f]; c += 1.f; }
    }
    centers[((size_t)b * K_ + k) * F_ + f] = s / fmaxf(c, 1.0f);
}

// ---------------------------------------------------------------------------
// 6) MLPs: kernel-gate (softmax -> attn f16) and bias-gate (biases f32)
// ---------------------------------------------------------------------------
__global__ __launch_bounds__(64)
void mlp_kernel(const float* __restrict__ centers,
                const float* __restrict__ kgw1, const float* __restrict__ kgb1,
                const float* __restrict__ kgw2, const float* __restrict__ kgb2,
                const float* __restrict__ kgw3, const float* __restrict__ kgb3,
                const float* __restrict__ bgw1, const float* __restrict__ bgb1,
                const float* __restrict__ bgw2, const float* __restrict__ bgb2,
                const float* __restrict__ bgw3, const float* __restrict__ bgb3,
                _Float16* __restrict__ attn, float* __restrict__ biases) {
    __shared__ float cf[F_];
    __shared__ float h1[HID_];
    __shared__ float h2[HID_];
    __shared__ float o3[HID_];
    const int row = blockIdx.x;       // b*K_ + k
    const int t = threadIdx.x;
    for (int i = t; i < F_; i += 64) cf[i] = centers[(size_t)row * F_ + i];
    __syncthreads();
    // ---- kernel-gate path ----
    float s = kgb1[t];
    for (int f = 0; f < F_; ++f) s += cf[f] * kgw1[f * HID_ + t];
    h1[t] = fmaxf(s, 0.f);
    __syncthreads();
    s = kgb2[t];
    for (int j = 0; j < HID_; ++j) s += h1[j] * kgw2[j * HID_ + t];
    h2[t] = fmaxf(s, 0.f);
    __syncthreads();
    if (t < NB_) {
        float v = kgb3[t];
        for (int j = 0; j < HID_; ++j) v += h2[j] * kgw3[j * NB_ + t];
        o3[t] = v;
    }
    __syncthreads();
    if (t == 0) {
        float mx = -3.0e38f;
        for (int i = 0; i < NB_; ++i) mx = fmaxf(mx, o3[i]);
        float sum = 0.f;
        for (int i = 0; i < NB_; ++i) { float e = expf(o3[i] - mx); o3[i] = e; sum += e; }
        float inv = 1.0f / sum;
        for (int i = 0; i < NB_; ++i) attn[(size_t)row * NB_ + i] = (_Float16)(o3[i] * inv);
    }
    __syncthreads();
    // ---- bias-gate path ----
    s = bgb1[t];
    for (int f = 0; f < F_; ++f) s += cf[f] * bgw1[f * HID_ + t];
    h1[t] = fmaxf(s, 0.f);
    __syncthreads();
    s = bgb2[t];
    for (int j = 0; j < HID_; ++j) s += h1[j] * bgw2[j * HID_ + t];
    h2[t] = fmaxf(s, 0.f);
    __syncthreads();
    s = bgb3[t];
    for (int j = 0; j < HID_; ++j) s += h2[j] * bgw3[j * HID_ + t];
    biases[(size_t)row * CO_ + t] = s;
}

// ---------------------------------------------------------------------------
// 7) pack base_kernels (NB,F,CO) f32 -> basep[col][n] f16, col = c*F_+f
//    (B-fragment friendly: for fixed output col, 32 n-values contiguous)
// ---------------------------------------------------------------------------
__global__ void pack_base(const float* __restrict__ base, _Float16* __restrict__ basep) {
    int t = blockIdx.x * blockDim.x + threadIdx.x;
    if (t >= NB_ * COLS_) return;
    int col = t >> 5, n = t & 31;
    int c = col / F_, f = col % F_;
    basep[t] = (_Float16)base[((size_t)n * F_ + f) * CO_ + c];
}

// ---------------------------------------------------------------------------
// 8) mixture GEMM: attn(64x32) @ base(32x18432) via v_wmma_f32_16x16x32_f16.
//    K=32 -> exactly one WMMA per 16x16 tile. Output written TRANSPOSED as
//    kerT[b][k][c*F_+f] f16 so the token GEMM's B loads are contiguous.
// ---------------------------------------------------------------------------
__global__ __launch_bounds__(32)
void mix_gemm(const _Float16* __restrict__ attn, const _Float16* __restrict__ basep,
              _Float16* __restrict__ kerT) {
    const int b = blockIdx.x, mt = blockIdx.y, cg = blockIdx.z;
    const int lane = threadIdx.x;
    const int m = lane & 15, g = lane >> 4;
    // A fragment: lane row = mt*16+m ; halves K = [g*8, g*8+8) and [16+g*8, +8)
    const _Float16* Ap = attn + (size_t)(b * K_ + mt * 16 + m) * NB_;
    v16h a;
    v8h* ap = reinterpret_cast<v8h*>(&a);
    ap[0] = *reinterpret_cast<const v8h*>(Ap + g * 8);
    ap[1] = *reinterpret_cast<const v8h*>(Ap + 16 + g * 8);
    for (int i = 0; i < 8; ++i) {
        int colt = cg * 8 + i;
        int col  = colt * 16 + m;                       // B fragment: N = lane%16
        const _Float16* Bp = basep + (size_t)col * NB_; // K halves = [g*8, g*8+8) and [16+g*8, +8)
        v16h bb;
        v8h* bp = reinterpret_cast<v8h*>(&bb);
        bp[0] = *reinterpret_cast<const v8h*>(Bp + g * 8);
        bp[1] = *reinterpret_cast<const v8h*>(Bp + 16 + g * 8);
        v8f acc = {};
        acc = __builtin_amdgcn_wmma_f32_16x16x32_f16(false, a, false, bb,
                                                     (short)0, acc, false, false);
#pragma unroll
        for (int r = 0; r < 8; ++r) {
            int krow = mt * 16 + g * 8 + r;             // D: rows r + g*8
            kerT[(size_t)(b * K_ + krow) * COLS_ + col] = (_Float16)acc[r];
        }
    }
}

// ---------------------------------------------------------------------------
// 9) grouped token GEMM: per (b, cluster k, 16-token tile): 4 waves x 16
//    channels, 9 k-chunks of 32 over F=288. A rows gathered via order list,
//    B rows contiguous in kerT, bias fused, scatter-store to NCHW out.
// ---------------------------------------------------------------------------
__global__ __launch_bounds__(128)
void tok_gemm(const _Float16* __restrict__ path, const _Float16* __restrict__ kerT,
              const float* __restrict__ biases, const int* __restrict__ order,
              const int* __restrict__ offs, const int* __restrict__ cnts,
              float* __restrict__ out) {
    const int b = blockIdx.x, k = blockIdx.y, mt = blockIdx.z;
    const int cnt = cnts[b * K_ + k];
    if (mt * 16 >= cnt) return;                 // block-uniform: EXEC stays full
    const int off = offs[b * K_ + k];
    const int wv = threadIdx.x >> 5;            // N-tile (channel group)
    const int lane = threadIdx.x & 31;
    const int m = lane & 15, g = lane >> 4;
    const int* ord = order + b * P_ + off;

    const int tpos = mt * 16 + m;
    const int tok  = (tpos < cnt) ? ord[tpos] : -1;
    const _Float16* Ap = path + ((size_t)b * P_ + (tok >= 0 ? tok : 0)) * F_;
    const int ch = wv * 16 + m;                 // output channel (= D column N)
    const _Float16* Bp = kerT + (size_t)(b * K_ + k) * COLS_ + (size_t)ch * F_;

    v8f acc = {};
    for (int kc = 0; kc < 9; ++kc) {
        const int f0 = kc * 32;
        v16h a;
        v8h* ap = reinterpret_cast<v8h*>(&a);
        if (tok >= 0) {
            ap[0] = *reinterpret_cast<const v8h*>(Ap + f0 + g * 8);
            ap[1] = *reinterpret_cast<const v8h*>(Ap + f0 + 16 + g * 8);
        } else {
            v8h z = {};
            ap[0] = z; ap[1] = z;
        }
        v16h bb;
        v8h* bp = reinterpret_cast<v8h*>(&bb);
        bp[0] = *reinterpret_cast<const v8h*>(Bp + f0 + g * 8);
        bp[1] = *reinterpret_cast<const v8h*>(Bp + f0 + 16 + g * 8);
        if (kc < 8) __builtin_prefetch(Bp + f0 + 32, 0, 3);   // global_prefetch_b8
        acc = __builtin_amdgcn_wmma_f32_16x16x32_f16(false, a, false, bb,
                                                     (short)0, acc, false, false);
    }
    const float bias = biases[(size_t)(b * K_ + k) * CO_ + ch];
#pragma unroll
    for (int r = 0; r < 8; ++r) {
        const int tp2 = mt * 16 + g * 8 + r;    // D: rows r + g*8
        if (tp2 < cnt) {
            const int tok2 = ord[tp2];
            out[((size_t)(b * CO_ + ch)) * P_ + tok2] = acc[r] + bias;
        }
    }
}

// ---------------------------------------------------------------------------
extern "C" void kernel_launch(void* const* d_in, const int* in_sizes, int n_in,
                              void* d_out, int out_size, void* d_ws, size_t ws_size,
                              hipStream_t stream) {
    const float* x    = (const float*)d_in[0];
    const float* base = (const float*)d_in[1];
    const float* kgw1 = (const float*)d_in[2];
    const float* kgb1 = (const float*)d_in[3];
    const float* kgw2 = (const float*)d_in[4];
    const float* kgb2 = (const float*)d_in[5];
    const float* kgw3 = (const float*)d_in[6];
    const float* kgb3 = (const float*)d_in[7];
    const float* bgw1 = (const float*)d_in[8];
    const float* bgb1 = (const float*)d_in[9];
    const float* bgw2 = (const float*)d_in[10];
    const float* bgb2 = (const float*)d_in[11];
    const float* bgw3 = (const float*)d_in[12];
    const float* bgb3 = (const float*)d_in[13];

    char* w = (char*)d_ws;
    size_t o = 0;
    auto carve = [&](size_t bytes) -> char* {
        char* p = w + o;
        o = (o + bytes + 255) & ~(size_t)255;
        return p;
    };
    float*    patf    = (float*)   carve((size_t)B_ * P_ * F_ * 4);     // 9.44 MB
    _Float16* path    = (_Float16*)carve((size_t)B_ * P_ * F_ * 2);     // 4.72 MB
    float*    obs     = (float*)   carve((size_t)B_ * P_ * C_ * 4);     // 1.05 MB
    float*    centers = (float*)   carve((size_t)B_ * K_ * F_ * 4);
    _Float16* attn    = (_Float16*)carve((size_t)B_ * K_ * NB_ * 2);
    float*    biases  = (float*)   carve((size_t)B_ * K_ * CO_ * 4);
    _Float16* basep   = (_Float16*)carve((size_t)NB_ * COLS_ * 2);      // 1.18 MB
    _Float16* kerT    = (_Float16*)carve((size_t)B_ * K_ * COLS_ * 2);  // 2.36 MB
    int*      order   = (int*)     carve((size_t)B_ * P_ * 4);
    int*      offs    = (int*)     carve((size_t)B_ * K_ * 4);
    int*      cnts    = (int*)     carve((size_t)B_ * K_ * 4);
    (void)ws_size; (void)in_sizes; (void)n_in; (void)out_size;

    float* outp = (float*)d_out;                         // (B, CO, H, W) f32
    int*   idxo = (int*)d_out + (size_t)B_ * CO_ * P_;   // (B, P) int32

    unfold_kernel<<<(B_ * P_ * F_ + 255) / 256, 256, 0, stream>>>(x, patf, path);
    obs_kernel<<<(B_ * P_ * C_ + 255) / 256, 256, 0, stream>>>(patf, obs);
    kmeans_kernel<<<B_, 1024, 0, stream>>>(obs, idxo);
    rank_kernel<<<B_, 1024, 0, stream>>>(idxo, order, offs, cnts);
    centers_kernel<<<dim3(B_, K_), F_, 0, stream>>>(idxo, patf, centers);
    mlp_kernel<<<B_ * K_, HID_, 0, stream>>>(centers,
        kgw1, kgb1, kgw2, kgb2, kgw3, kgb3,
        bgw1, bgb1, bgw2, bgb2, bgw3, bgb3, attn, biases);
    pack_base<<<(NB_ * COLS_ + 255) / 256, 256, 0, stream>>>(base, basep);
    mix_gemm<<<dim3(B_, 2, COLS_ / 16 / 8), 32, 0, stream>>>(attn, basep, kerT);
    tok_gemm<<<dim3(B_, K_, P_ / 16), 128, 0, stream>>>(path, kerT, biases,
                                                        order, offs, cnts, outp);
}
